// Core_11570641896180
// MI455X (gfx1250) — matrix operand order repaired and checked by reading, not compiled, not run on hardware
//
#include <hip/hip_runtime.h>
#include <cstdint>
#include <cstddef>

// Problem constants (match reference)
#define TT     464
#define HH     256
#define BB     64
#define FOURH  1024
#define THK    (TT * HH)      // 118784 = flattened K of trailing dense
#define MROWS  (TT * BB)      // 29696  = rows of the hoisted X@W GEMM
#define KSPLIT 1024
#define SPLITS (THK / KSPLIT) // 116
#define NWG_LSTM 16           // persistent workgroups for the recurrence

typedef __attribute__((ext_vector_type(8)))  __bf16 v8bf;
typedef __attribute__((ext_vector_type(16))) __bf16 v16bf;
typedef __attribute__((ext_vector_type(8)))  float  v8f;
typedef __attribute__((ext_vector_type(4)))  float  v4f;

// ---------------------------------------------------------------------------
// WMMA fragment loaders (wave32, v_wmma_f32_16x16x32_bf16)
// A 16x32 (row-major, stride ld elems): lane r=lane&15 row, half=lane>>4:
//   elems 0..7  -> K = k0 + 8*half + e
//   elems 8..15 -> K = k0 + 16 + 8*half + (e-8)
// B 32x16 loaded from a TRANSPOSED [N][K] bf16 array (stride ld elems):
//   lane r = column N, same K mapping as A (8*half + e, 16 + 8*half + (e-8))
// ---------------------------------------------------------------------------
__device__ __forceinline__ v16bf load_fragA(const __bf16* base, int ld,
                                            int row0, int k0, int r, int half) {
  const __bf16* p = base + (size_t)(row0 + r) * (size_t)ld + k0 + half * 8;
  v8bf a0 = *(const v8bf*)p;
  v8bf a1 = *(const v8bf*)(p + 16);
  v16bf A;
#pragma unroll
  for (int i = 0; i < 8; ++i) { A[i] = a0[i]; A[i + 8] = a1[i]; }
  return A;
}

__device__ __forceinline__ v16bf load_fragB(const __bf16* baseT, int ld,
                                            int n0, int k0, int r, int half) {
  const __bf16* p = baseT + (size_t)(n0 + r) * (size_t)ld + k0 + half * 8;
  v8bf b0 = *(const v8bf*)p;
  v8bf b1 = *(const v8bf*)(p + 16);
  v16bf B;
#pragma unroll
  for (int i = 0; i < 8; ++i) { B[i] = b0[i]; B[i + 8] = b1[i]; }
  return B;
}

__device__ __forceinline__ v8f wmma_bf16(v16bf A, v16bf B, v8f C) {
  return __builtin_amdgcn_wmma_f32_16x16x32_bf16(false, A, false, B,
                                                 (short)0, C, false, false);
}

__device__ __forceinline__ float sigmoidf_(float x) {
  return 1.0f / (1.0f + __expf(-x));
}

// ---------------------------------------------------------------------------
// Convert feature_encoded [B,T,H] f32 -> Xb [T*B, H] bf16 with row r' = t*B+b
// ---------------------------------------------------------------------------
__global__ __launch_bounds__(256) void k_convert_x(const float* __restrict__ x,
                                                   __bf16* __restrict__ xb) {
  int rp = blockIdx.x;            // 0..MROWS-1, r' = t*B + b
  int h  = threadIdx.x;
  int b  = rp & (BB - 1);
  int t  = rp >> 6;
  float v = x[((size_t)b * TT + t) * HH + h];
  xb[(size_t)rp * HH + h] = (__bf16)v;
}

// ---------------------------------------------------------------------------
// Tiled transpose + f32->bf16: src f32 [R][C] -> dst bf16 [C][R]
// ---------------------------------------------------------------------------
__global__ __launch_bounds__(256) void k_transpose_bf16(const float* __restrict__ src,
                                                        __bf16* __restrict__ dst,
                                                        int R, int C) {
  __shared__ float tile[32][33];
  int tx = threadIdx.x & 31;
  int ty = threadIdx.x >> 5;            // 0..7
  int c0 = blockIdx.x * 32;
  int r0 = blockIdx.y * 32;
#pragma unroll
  for (int i = 0; i < 4; ++i) {
    int row = r0 + ty + i * 8;
    tile[ty + i * 8][tx] = src[(size_t)row * C + c0 + tx];
  }
  __syncthreads();
#pragma unroll
  for (int i = 0; i < 4; ++i) {
    int crow = c0 + ty + i * 8;
    dst[(size_t)crow * R + r0 + tx] = (__bf16)tile[tx][ty + i * 8];
  }
}

// ---------------------------------------------------------------------------
// Init: h0 -> bf16 double buffer slot 0; zero the global barrier counter
// ---------------------------------------------------------------------------
__global__ __launch_bounds__(256) void k_init_h(const float* __restrict__ h0,
                                                __bf16* __restrict__ hb16,
                                                unsigned int* __restrict__ ctr) {
  int idx = blockIdx.x * 256 + threadIdx.x;
  hb16[idx] = (__bf16)h0[idx];
  if (idx == 0) *ctr = 0u;
}

// ---------------------------------------------------------------------------
// GEMM1: XWp[t][n][b] = (Xb[t*B+b,:] @ W)[n] + bias[n]   (bf16 WMMA, f32 acc)
// Output layout [T][4H][B] so the recurrence reads each lane's accumulator
// init as two contiguous b128 loads.
// grid (MROWS/16, 4) x 128 threads; wave handles 1 M-tile x 4 N-tiles
// ---------------------------------------------------------------------------
__global__ __launch_bounds__(128) void k_gemm1(const __bf16* __restrict__ Xb,
                                               const __bf16* __restrict__ WT,   // [4H][H]
                                               const float*  __restrict__ bias, // [4H]
                                               float* __restrict__ XWp) {       // [T][4H][B]
  int tid  = threadIdx.x;
  int wid  = tid >> 5;
  int lane = tid & 31;
  int r    = lane & 15;
  int half = lane >> 4;
  int row0 = blockIdx.x * 16;          // rows r' = t*64+b ; 16 | 64 so t const
  int t    = row0 >> 6;
  int b0   = (row0 & 63) + half * 8;   // lane's first batch index
  int nbas = blockIdx.y * 256 + wid * 64;

  v8f acc[4];
#pragma unroll
  for (int j = 0; j < 4; ++j) acc[j] = (v8f)(0.0f);

  for (int kk = 0; kk < HH / 32; ++kk) {
    int k0 = kk * 32;
    v16bf A = load_fragA(Xb, HH, row0, k0, r, half);
#pragma unroll
    for (int j = 0; j < 4; ++j) {
      v16bf B = load_fragB(WT, HH, nbas + j * 16, k0, r, half);
      acc[j] = wmma_bf16(A, B, acc[j]);
    }
  }
#pragma unroll
  for (int j = 0; j < 4; ++j) {
    int n = nbas + j * 16 + r;
    float bv = bias[n];
    v4f lo = { acc[j][0] + bv, acc[j][1] + bv, acc[j][2] + bv, acc[j][3] + bv };
    v4f hi = { acc[j][4] + bv, acc[j][5] + bv, acc[j][6] + bv, acc[j][7] + bv };
    float* dst = XWp + ((size_t)t * FOURH + n) * BB + b0;
    __builtin_nontemporal_store(lo, (v4f*)dst);
    __builtin_nontemporal_store(hi, (v4f*)(dst + 4));
  }
}

// ---------------------------------------------------------------------------
// Persistent LSTM recurrence: 16 WGs x 4 waves = 64 waves. All 4 waves of a
// workgroup share one h-column tile (hcol = blockIdx.x) and differ in batch
// tile (m = wid), so the workgroup's 32KB slice of U^T (4 gates x 16 cols x
// 256 K, bf16) is staged into LDS ONCE and served by ds_load_b128 for all 464
// steps. c-state lives in registers across the whole sequence; h passes
// through a double-buffered global bf16 buffer; agent-scope barrier per step;
// next step's XW block is prefetched during compute.
// ---------------------------------------------------------------------------
__global__ __launch_bounds__(128) void k_lstm(const float*  __restrict__ XWp,   // [T][4H][B]
                                              const __bf16* __restrict__ UT,    // [4H][H]
                                              const float*  __restrict__ c0g,   // [B][H]
                                              __bf16* __restrict__ hb16,        // [2][B*H]
                                              __bf16* __restrict__ seqb,        // [B][T*H]
                                              float* __restrict__ outH,         // [B][H]
                                              float* __restrict__ outC,         // [B][H]
                                              unsigned int* __restrict__ ctr) {
  __shared__ __bf16 Us[4 * 16 * 256];   // 32 KB: rows (g,n) -> g*16+n, ld=256
  int tid  = threadIdx.x;
  int wid  = tid >> 5;
  int lane = tid & 31;
  int r    = lane & 15;
  int half = lane >> 4;
  int m    = wid;                      // batch tile (0..3)
  int hcol = blockIdx.x;               // h column tile (0..15)
  int row0 = m * 16;
  int hidx = hcol * 16 + r;

  // Stage U^T slice for this workgroup's gate columns into LDS.
  // 64 rows x 256 bf16 = 16384 elems; 128 threads x 16 chunks of v8bf.
#pragma unroll
  for (int i = 0; i < 16; ++i) {
    int idx  = i * 1024 + tid * 8;     // element index
    int lrow = idx >> 8;               // 0..63  (= g*16 + n)
    int k    = idx & 255;
    int g    = lrow >> 4;
    int n    = lrow & 15;
    *(v8bf*)&Us[(size_t)lrow * 256 + k] =
        *(const v8bf*)&UT[((size_t)g * 256 + hcol * 16 + n) * HH + k];
  }
  __syncthreads();

  // Load persistent cell-state fragment
  v8f c;
#pragma unroll
  for (int e = 0; e < 8; ++e) {
    int b = row0 + e + half * 8;
    c[e] = c0g[(size_t)b * HH + hidx];
  }

  for (int t = 0; t < TT; ++t) {
    int p = t & 1;
    const __bf16* hsrc = hb16 + (size_t)p * (BB * HH);
    __bf16*       hdst = hb16 + (size_t)(1 - p) * (BB * HH);

    // Prefetch next step's XW block (4 gates x 16 cols x 64 batch = 16KB/wg)
    if (t + 1 < TT) {
#pragma unroll
      for (int g = 0; g < 4; ++g) {
        const float* pf = XWp + ((size_t)(t + 1) * FOURH + g * 256 + hcol * 16) * BB
                        + lane * 32;   // 32 lanes x 128B = 4KB per gate
        __builtin_prefetch(pf, 0, 1);
      }
    }

    // acc <- precomputed x_t @ W + bias (two b128 loads per gate)
    v8f acc[4];
#pragma unroll
    for (int g = 0; g < 4; ++g) {
      const float* src = XWp + ((size_t)t * FOURH + g * 256 + hidx) * BB
                       + row0 + half * 8;
      v4f lo = *(const v4f*)src;
      v4f hi = *(const v4f*)(src + 4);
#pragma unroll
      for (int e = 0; e < 4; ++e) { acc[g][e] = lo[e]; acc[g][e + 4] = hi[e]; }
    }
    // acc += h_{t-1} @ U   (K = 256, B fragments served from LDS)
#pragma unroll
    for (int kk = 0; kk < HH / 32; ++kk) {
      int k0 = kk * 32;
      v16bf A = load_fragA(hsrc, HH, row0, k0, r, half);
#pragma unroll
      for (int g = 0; g < 4; ++g) {
        v16bf B = load_fragB(Us, 256, g * 16, k0, r, half);
        acc[g] = wmma_bf16(A, B, acc[g]);
      }
    }
    // Gates: i,f,g,o ; c' = f*c + i*relu(g) ; h = o*relu(c')
    v8f hn;
#pragma unroll
    for (int e = 0; e < 8; ++e) {
      float ig = sigmoidf_(acc[0][e]);
      float fg = sigmoidf_(acc[1][e]);
      float gg = acc[2][e] > 0.0f ? acc[2][e] : 0.0f;
      float og = sigmoidf_(acc[3][e]);
      float cn = fg * c[e] + ig * gg;
      c[e] = cn;
      hn[e] = og * (cn > 0.0f ? cn : 0.0f);
    }
    // Store h (next-step operand + sequence output)
#pragma unroll
    for (int e = 0; e < 8; ++e) {
      int b = row0 + e + half * 8;
      __bf16 hb = (__bf16)hn[e];
      hdst[(size_t)b * HH + hidx] = hb;
      seqb[(size_t)b * THK + (size_t)t * HH + hidx] = hb;
    }
    if (t == TT - 1) {
#pragma unroll
      for (int e = 0; e < 8; ++e) {
        int b = row0 + e + half * 8;
        outH[(size_t)b * HH + hidx] = hn[e];
        outC[(size_t)b * HH + hidx] = c[e];
      }
    }
    // Device-wide step barrier (monotonic counter; release/acquire)
    __threadfence();
    __syncthreads();
    if (tid == 0) {
      __hip_atomic_fetch_add(ctr, 1u, __ATOMIC_RELEASE, __HIP_MEMORY_SCOPE_AGENT);
      unsigned int target = (unsigned int)NWG_LSTM * (unsigned int)(t + 1);
      while (__hip_atomic_load(ctr, __ATOMIC_ACQUIRE, __HIP_MEMORY_SCOPE_AGENT) < target)
        __builtin_amdgcn_s_sleep(2);
    }
    __syncthreads();
  }
}

// ---------------------------------------------------------------------------
// Dense split-K: part[s][b][n] = seqb[b, sK..sK+1024) @ WdT[n, sK..sK+1024)
// grid (4, 4, SPLITS) x 128 threads; one 16x16 tile per wave
// ---------------------------------------------------------------------------
__global__ __launch_bounds__(128) void k_dense(const __bf16* __restrict__ seqb, // [B][THK]
                                               const __bf16* __restrict__ WdT,  // [H][THK]
                                               float* __restrict__ part) {      // [SPLITS][B][H]
  int tid  = threadIdx.x;
  int wid  = tid >> 5;
  int lane = tid & 31;
  int r    = lane & 15;
  int half = lane >> 4;
  int row0 = blockIdx.x * 16;                 // batch tile
  int n0   = blockIdx.y * 64 + wid * 16;      // output column tile
  int s    = blockIdx.z;

  v8f acc = (v8f)(0.0f);
  for (int kk = 0; kk < KSPLIT / 32; ++kk) {
    int k0 = s * KSPLIT + kk * 32;
    v16bf A = load_fragA(seqb, THK, row0, k0, r, half);
    v16bf B = load_fragB(WdT, THK, n0, k0, r, half);
    acc = wmma_bf16(A, B, acc);
  }
  int n = n0 + r;
#pragma unroll
  for (int e = 0; e < 8; ++e) {
    int b = row0 + e + half * 8;
    __builtin_nontemporal_store(acc[e],
        &part[((size_t)s * BB + b) * HH + n]);
  }
}

// ---------------------------------------------------------------------------
// Deterministic split-K reduction + bias + relu -> core_output
// ---------------------------------------------------------------------------
__global__ __launch_bounds__(256) void k_reduce(const float* __restrict__ part,
                                                const float* __restrict__ bd,
                                                float* __restrict__ out) {
  int idx = blockIdx.x * 256 + threadIdx.x;   // 0..B*H-1
  int n = idx & (HH - 1);
  float s = bd[n];
  for (int sp = 0; sp < SPLITS; ++sp)
    s += part[(size_t)sp * (BB * HH) + idx];
  out[idx] = s > 0.0f ? s : 0.0f;
}

// ---------------------------------------------------------------------------
// Host launcher
// ---------------------------------------------------------------------------
extern "C" void kernel_launch(void* const* d_in, const int* in_sizes, int n_in,
                              void* d_out, int out_size, void* d_ws, size_t ws_size,
                              hipStream_t stream) {
  const float* feat = (const float*)d_in[0]; // [B,T,H]
  const float* h0   = (const float*)d_in[1]; // [B,H]
  const float* c0   = (const float*)d_in[2]; // [B,H]
  const float* W    = (const float*)d_in[3]; // [H,4H]
  const float* U    = (const float*)d_in[4]; // [H,4H]
  const float* bi   = (const float*)d_in[5]; // [4H]
  const float* Wd   = (const float*)d_in[6]; // [T*H,H]
  const float* bd   = (const float*)d_in[7]; // [H]

  float* out_core = (float*)d_out;
  float* out_hT   = out_core + BB * HH;
  float* out_cT   = out_hT   + BB * HH;

  // Workspace carve-up (256-byte aligned sections)
  char* ws = (char*)d_ws;
  size_t off = 0;
  auto carve = [&](size_t bytes) -> char* {
    char* p = ws + off;
    off = (off + bytes + 255) & ~(size_t)255;
    return p;
  };
  float*  XWp  = (float*) carve((size_t)MROWS * FOURH * 4); // 121.6 MB
  __bf16* WdT  = (__bf16*)carve((size_t)HH * THK * 2);      //  60.8 MB
  __bf16* Xb   = (__bf16*)carve((size_t)MROWS * HH * 2);    //  15.2 MB
  __bf16* seqb = (__bf16*)carve((size_t)BB * THK * 2);      //  15.2 MB
  float*  part = (float*) carve((size_t)SPLITS * BB * HH * 4); // 7.6 MB
  __bf16* WT   = (__bf16*)carve((size_t)FOURH * HH * 2);
  __bf16* UT   = (__bf16*)carve((size_t)FOURH * HH * 2);
  __bf16* hb16 = (__bf16*)carve((size_t)2 * BB * HH * 2);
  unsigned int* ctr = (unsigned int*)carve(256);
  (void)ws_size; (void)in_sizes; (void)n_in; (void)out_size;

  // 1) format conversions / transposes
  k_convert_x<<<dim3(MROWS), 256, 0, stream>>>(feat, Xb);
  k_transpose_bf16<<<dim3(FOURH / 32, HH / 32), 256, 0, stream>>>(W,  WT,  HH, FOURH);
  k_transpose_bf16<<<dim3(FOURH / 32, HH / 32), 256, 0, stream>>>(U,  UT,  HH, FOURH);
  k_transpose_bf16<<<dim3(HH / 32, THK / 32), 256, 0, stream>>>(Wd, WdT, THK, HH);
  k_init_h<<<dim3((BB * HH) / 256), 256, 0, stream>>>(h0, hb16, ctr);

  // 2) hoisted input projection X @ W + b  (writes [T][4H][B] layout)
  k_gemm1<<<dim3(MROWS / 16, 4), 128, 0, stream>>>(Xb, WT, bi, XWp);

  // 3) persistent recurrence
  k_lstm<<<dim3(NWG_LSTM), 128, 0, stream>>>(XWp, UT, c0, hb16, seqb,
                                             out_hT, out_cT, ctr);

  // 4) trailing dense (split-K) + deterministic reduction
  k_dense<<<dim3(BB / 16, 4, SPLITS), 128, 0, stream>>>(seqb, WdT, part);
  k_reduce<<<dim3((BB * HH) / 256), 256, 0, stream>>>(part, bd, out_core);
}
